// Conv1dAttention_61778809585928
// MI455X (gfx1250) — compile-verified, matrix operand order checked by reading, not run on hardware
//
#include <hip/hip_runtime.h>

// ---------------------------------------------------------------------------
// Conv1dAttention for MI455X (gfx1250, wave32, WMMA bf16 16x16x32).
//
//  K1 (x4, templated): conv1d(K=3,pad=1)+bias+BN+LeakyReLU.
//     q -> Qb  bf16 [B][L][C]   (A-frag friendly: rows l, contiguous c)
//     k -> Kb  bf16 [B][L][C]   (B-frag friendly: cols, contiguous c)
//     v -> Vt  bf16 [B][C][L]   (A-frag friendly for P@V: rows c, contig j)
//     pe-> PE  f32  [B][C][L]
//  K2a: per attn row j, full-row softmax stats m_j, Z_j (reference einsum
//       contracts over the softmax ROW index -> stats must be precomputed).
//       Double-buffered K fragments so loads overlap WMMA + softmax VALU.
//  K2b: per 16-wide l tile: recompute S[j][l] with WMMA (Q frags double-
//       buffered, V frags hoisted per chunk), P = exp(S-m_j)/Z_j, transpose
//       P via LDS, second WMMA GEMM out[c][l] += Vt[c][j]*P[j][l], add PE.
//
// Workspace: Qb 8MB | Kb 8MB | Vt 8MB | PE 16MB | Mst 128KB | Zst 128KB
//            = ~40.3 MB required in d_ws.
// ---------------------------------------------------------------------------

#define BATCH 16
#define CIN   64
#define COUT  128
#define LEN   2048
#define LT    128   // conv l-tile per block

typedef __attribute__((ext_vector_type(16))) __bf16 v16bf;
typedef __attribute__((ext_vector_type(8)))  float  v8f;

__device__ __forceinline__ unsigned short f2bf(float f) {
  unsigned u = __float_as_uint(f);
  u += 0x7fffu + ((u >> 16) & 1u);          // round-to-nearest-even
  return (unsigned short)(u >> 16);
}

// ---------------------------------------------------------------------------
// Kernel 1: fused conv + bias + BN + LeakyReLU.  One block = (b, 128-l tile).
// ---------------------------------------------------------------------------
template <int P>
__global__ __launch_bounds__(256) void conv_bn_lrelu_kernel(
    const float* __restrict__ x, const float* __restrict__ w,
    const float* __restrict__ bias, const float* __restrict__ gamma,
    const float* __restrict__ beta, const float* __restrict__ mean,
    const float* __restrict__ var, unsigned short* __restrict__ obf,
    float* __restrict__ of32) {
  __shared__ float xs[CIN][LT + 2];
  const int b = blockIdx.y;
  const int lbase = blockIdx.x * LT;
  const int tid = threadIdx.x;

  for (int idx = tid; idx < CIN * (LT + 2); idx += 256) {
    int ci = idx / (LT + 2), ll = idx % (LT + 2);
    int gl = lbase + ll - 1;
    xs[ci][ll] = (gl >= 0 && gl < LEN) ? x[((size_t)b * CIN + ci) * LEN + gl]
                                       : 0.f;
  }
  __syncthreads();

  for (int it = 0; it < 4; ++it) {                 // 1024 (cout,lg) slots / 256
    int slot = it * 256 + tid;
    int cout = slot & (COUT - 1);
    int lg = slot >> 7;                            // 0..7 (16 l's each)
    const int lo = lg * 16;
    float acc[16];
#pragma unroll
    for (int i = 0; i < 16; ++i) acc[i] = 0.f;
    const float* wr = w + (size_t)cout * CIN * 3;
    for (int ci = 0; ci < CIN; ++ci) {
      float w0 = wr[ci * 3 + 0], w1 = wr[ci * 3 + 1], w2 = wr[ci * 3 + 2];
      const float* xr = &xs[ci][lo];
      float xm = xr[0], xc = xr[1];
#pragma unroll
      for (int i = 0; i < 16; ++i) {
        float xp = xr[i + 2];
        acc[i] = fmaf(w0, xm, fmaf(w1, xc, fmaf(w2, xp, acc[i])));
        xm = xc; xc = xp;
      }
    }
    float sc = gamma[cout] * rsqrtf(var[cout] + 1e-5f);
    float sh = beta[cout] - mean[cout] * sc;
    float bi = bias[cout];
#pragma unroll
    for (int i = 0; i < 16; ++i) {
      float y = (acc[i] + bi) * sc + sh;
      y = y > 0.f ? y : 0.3f * y;
      int l = lbase + lo + i;
      if (P == 0 || P == 1) {
        obf[((size_t)b * LEN + l) * COUT + cout] = f2bf(y);
      } else if (P == 2) {
        obf[((size_t)b * COUT + cout) * LEN + l] = f2bf(y);
      } else {
        of32[((size_t)b * COUT + cout) * LEN + l] = y;
      }
    }
  }
}

// ---------------------------------------------------------------------------
// Kernel 2a: softmax row stats; double-buffered K fragments.
// One wave per 16 attn rows.  D-frag: vgpr r, lanes0-15 row r, 16-31 row r+8.
// ---------------------------------------------------------------------------
__global__ __launch_bounds__(32) void attn_stats_kernel(
    const unsigned short* __restrict__ Qb, const unsigned short* __restrict__ Kb,
    float* __restrict__ Mst, float* __restrict__ Zst) {
  const int b = blockIdx.y;
  const int i0 = blockIdx.x * 16;
  const int t = threadIdx.x, lane16 = t & 15, half = t >> 4;

  v16bf qa[4];
  const unsigned short* qrow = Qb + ((size_t)b * LEN + (i0 + lane16)) * COUT;
#pragma unroll
  for (int kc = 0; kc < 4; ++kc)
    qa[kc] = *(const v16bf*)(qrow + kc * 32 + half * 16);

  float mrun[8], lrun[8];
#pragma unroll
  for (int r = 0; r < 8; ++r) { mrun[r] = -1e30f; lrun[r] = 0.f; }

  auto loadK = [&](v16bf* dst, int j0) {
#pragma unroll
    for (int jt = 0; jt < 2; ++jt)
#pragma unroll
      for (int kc = 0; kc < 4; ++kc)
        dst[jt * 4 + kc] = *(const v16bf*)(
            Kb + ((size_t)b * LEN + (j0 + jt * 16 + lane16)) * COUT +
            kc * 32 + half * 16);
  };
  auto procChunk = [&](const v16bf* kf) {
    v8f s0 = {}; v8f s1 = {};
#pragma unroll
    for (int kc = 0; kc < 4; ++kc) {
      s0 = __builtin_amdgcn_wmma_f32_16x16x32_bf16(false, qa[kc], false, kf[kc],
                                                   (short)0, s0, false, false);
      s1 = __builtin_amdgcn_wmma_f32_16x16x32_bf16(false, qa[kc], false,
                                                   kf[4 + kc], (short)0, s1,
                                                   false, false);
    }
#pragma unroll
    for (int r = 0; r < 8; ++r) {
      float mx = fmaxf(s0[r], s1[r]);
#pragma unroll
      for (int off = 1; off < 16; off <<= 1)
        mx = fmaxf(mx, __shfl_xor(mx, off, 32));
      float mnew = fmaxf(mrun[r], mx);
      float ssum = __expf(s0[r] - mnew) + __expf(s1[r] - mnew);
#pragma unroll
      for (int off = 1; off < 16; off <<= 1) ssum += __shfl_xor(ssum, off, 32);
      lrun[r] = lrun[r] * __expf(mrun[r] - mnew) + ssum;
      mrun[r] = mnew;
    }
  };

  v16bf kA[8], kB[8];
  loadK(kA, 0);
  for (int j0 = 0; j0 < LEN; j0 += 64) {           // 2x unrolled: no frag copies
    loadK(kB, j0 + 32);                            // overlap with chunk A work
    procChunk(kA);
    if (j0 + 64 < LEN) loadK(kA, j0 + 64);         // overlap with chunk B work
    procChunk(kB);
  }

  if (lane16 == 0) {
#pragma unroll
    for (int r = 0; r < 8; ++r) {
      int row = i0 + r + 8 * half;
      Mst[(size_t)b * LEN + row] = mrun[r];
      Zst[(size_t)b * LEN + row] = lrun[r];
    }
  }
}

// ---------------------------------------------------------------------------
// Kernel 2b: out[c][l] = sum_j Vt[c][j] * exp(S[j][l]-m_j)/Z_j  + PE[c][l].
// Q fragments double-buffered across chunks; V fragments hoisted per chunk.
// ---------------------------------------------------------------------------
__global__ __launch_bounds__(32) void attn_out_kernel(
    const unsigned short* __restrict__ Qb, const unsigned short* __restrict__ Kb,
    const unsigned short* __restrict__ Vt, const float* __restrict__ PE,
    const float* __restrict__ Mst, const float* __restrict__ Zst,
    float* __restrict__ out) {
  const int b = blockIdx.y;
  const int l0 = blockIdx.x * 16;
  const int t = threadIdx.x, lane16 = t & 15, half = t >> 4;

  __shared__ unsigned short pShm[16 * 32];   // P^T: [l local][j local]

  v16bf kb[4];                               // B-frag for fixed l columns
  const unsigned short* krow = Kb + ((size_t)b * LEN + (l0 + lane16)) * COUT;
#pragma unroll
  for (int kc = 0; kc < 4; ++kc)
    kb[kc] = *(const v16bf*)(krow + kc * 32 + half * 16);

  v8f acc[8];
#pragma unroll
  for (int cc = 0; cc < 8; ++cc) { v8f z = {}; acc[cc] = z; }

  auto loadQ = [&](v16bf* dst, int j0) {
#pragma unroll
    for (int jt = 0; jt < 2; ++jt)
#pragma unroll
      for (int kc = 0; kc < 4; ++kc)
        dst[jt * 4 + kc] = *(const v16bf*)(
            Qb + ((size_t)b * LEN + (j0 + jt * 16 + lane16)) * COUT +
            kc * 32 + half * 16);
  };

  auto procChunk = [&](const v16bf* qf, int j0) {
    // hoist V fragments: issue before GEMM1 so they overlap GEMM1+softmax
    v16bf vf[8];
#pragma unroll
    for (int cc = 0; cc < 8; ++cc)
      vf[cc] = *(const v16bf*)(
          Vt + ((size_t)b * COUT + cc * 16 + lane16) * LEN + j0 + half * 16);
    // ---- GEMM1: S[j][l], rows = j chunk (2 subtiles), cols = l tile
    v8f s0 = {}; v8f s1 = {};
#pragma unroll
    for (int kc = 0; kc < 4; ++kc) {
      s0 = __builtin_amdgcn_wmma_f32_16x16x32_bf16(false, qf[kc], false, kb[kc],
                                                   (short)0, s0, false, false);
      s1 = __builtin_amdgcn_wmma_f32_16x16x32_bf16(false, qf[4 + kc], false,
                                                   kb[kc], (short)0, s1,
                                                   false, false);
    }
    // ---- normalize with row-j stats, write P^T into LDS
#pragma unroll
    for (int r = 0; r < 8; ++r) {
      int jl0 = r + 8 * half;                     // j local for s0 tile
      float m0 = Mst[(size_t)b * LEN + j0 + jl0];
      float z0 = Zst[(size_t)b * LEN + j0 + jl0];
      pShm[lane16 * 32 + jl0] = f2bf(__expf(s0[r] - m0) / z0);
      int jl1 = 16 + jl0;                         // j local for s1 tile
      float m1 = Mst[(size_t)b * LEN + j0 + jl1];
      float z1 = Zst[(size_t)b * LEN + j0 + jl1];
      pShm[lane16 * 32 + jl1] = f2bf(__expf(s1[r] - m1) / z1);
    }
    __syncthreads();
    // ---- GEMM2: acc[c][l] += Vt[c][j] * P[j][l]
    v16bf pb = *(const v16bf*)(pShm + lane16 * 32 + half * 16);
#pragma unroll
    for (int cc = 0; cc < 8; ++cc)
      acc[cc] = __builtin_amdgcn_wmma_f32_16x16x32_bf16(false, vf[cc], false,
                                                        pb, (short)0, acc[cc],
                                                        false, false);
    __syncthreads();
  };

  v16bf qA[8], qB[8];
  loadQ(qA, 0);
  for (int j0 = 0; j0 < LEN; j0 += 64) {           // 2x unrolled double buffer
    loadQ(qB, j0 + 32);
    procChunk(qA, j0);
    if (j0 + 64 < LEN) loadQ(qA, j0 + 64);
    procChunk(qB, j0 + 32);
  }

  // ---- epilogue: D rows = c local (r+8*half), cols = l (lane16); add PE
#pragma unroll
  for (int cc = 0; cc < 8; ++cc) {
#pragma unroll
    for (int r = 0; r < 8; ++r) {
      int c = cc * 16 + r + 8 * half;
      size_t idx = ((size_t)b * COUT + c) * LEN + l0 + lane16;
      out[idx] = acc[cc][r] + PE[idx];
    }
  }
}

// ---------------------------------------------------------------------------
extern "C" void kernel_launch(void* const* d_in, const int* in_sizes, int n_in,
                              void* d_out, int out_size, void* d_ws,
                              size_t ws_size, hipStream_t stream) {
  const float* x = (const float*)d_in[0];
  // d_in order: x, then per proj (q,k,v,pe): w, b, gamma, beta, mean, var
  auto in = [&](int p, int j) { return (const float*)d_in[1 + p * 6 + j]; };

  char* ws = (char*)d_ws;
  unsigned short* Qb = (unsigned short*)(ws);                   //  8 MB
  unsigned short* Kb = (unsigned short*)(ws + (size_t)8  * 1048576);
  unsigned short* Vt = (unsigned short*)(ws + (size_t)16 * 1048576);
  float* PE  = (float*)(ws + (size_t)24 * 1048576);             // 16 MB
  float* Mst = (float*)(ws + (size_t)40 * 1048576);             // 128 KB
  float* Zst = (float*)(ws + (size_t)40 * 1048576 + 131072);    // 128 KB
  float* out = (float*)d_out;

  dim3 cgrid(LEN / LT, BATCH);
  conv_bn_lrelu_kernel<0><<<cgrid, 256, 0, stream>>>(
      x, in(0, 0), in(0, 1), in(0, 2), in(0, 3), in(0, 4), in(0, 5), Qb, nullptr);
  conv_bn_lrelu_kernel<1><<<cgrid, 256, 0, stream>>>(
      x, in(1, 0), in(1, 1), in(1, 2), in(1, 3), in(1, 4), in(1, 5), Kb, nullptr);
  conv_bn_lrelu_kernel<2><<<cgrid, 256, 0, stream>>>(
      x, in(2, 0), in(2, 1), in(2, 2), in(2, 3), in(2, 4), in(2, 5), Vt, nullptr);
  conv_bn_lrelu_kernel<3><<<cgrid, 256, 0, stream>>>(
      x, in(3, 0), in(3, 1), in(3, 2), in(3, 3), in(3, 4), in(3, 5), nullptr, PE);

  dim3 agrid(LEN / 16, BATCH);
  attn_stats_kernel<<<agrid, 32, 0, stream>>>(Qb, Kb, Mst, Zst);
  attn_out_kernel<<<agrid, 32, 0, stream>>>(Qb, Kb, Vt, PE, Mst, Zst, out);
}
